// CRF_44813688766602
// MI455X (gfx1250) — matrix operand order repaired and checked by reading, not compiled, not run on hardware
//
#include <hip/hip_runtime.h>
#include <hip/hip_bf16.h>

#define BATCH 4096
#define SEQ   512
#define NTAG  25
#define WAVES 4   // waves per block in forward kernel

#define LOG2E 1.4426950408889634f
#define LN2   0.6931471805599453f
#define NEGBIG (-1.0e30f)   // finite -inf stand-in: exp2 flushes to 0, fma-safe

typedef __attribute__((ext_vector_type(16))) _Float16 v16h;
typedef __attribute__((ext_vector_type(8)))  _Float16 v8h;
typedef __attribute__((ext_vector_type(8)))  float    v8f;

// DPP butterfly helpers: reductions within each 16-lane group, pure VALU
// (v_mov_dpp + v_max/v_add), no LDS-swizzle latency on the critical chain.
// quad_perm{1,0,3,2}=0xB1 (xor1), quad_perm{2,3,0,1}=0x4E (xor2),
// row_half_mirror=0x141 (covers 8-groups), row_mirror=0x140 (covers 16-groups).
template <int CTRL>
__device__ __forceinline__ float dpp_max(float x) {
    int m = __builtin_amdgcn_update_dpp(0, __float_as_int(x), CTRL, 0xF, 0xF, true);
    return fmaxf(x, __int_as_float(m));
}
template <int CTRL>
__device__ __forceinline__ float dpp_add(float x) {
    int m = __builtin_amdgcn_update_dpp(0, __float_as_int(x), CTRL, 0xF, 0xF, true);
    return x + __int_as_float(m);
}
__device__ __forceinline__ float red16_max(float x) {
    x = dpp_max<0xB1>(x); x = dpp_max<0x4E>(x);
    x = dpp_max<0x141>(x); x = dpp_max<0x140>(x);
    return x;
}
__device__ __forceinline__ float red16_add(float x) {
    x = dpp_add<0xB1>(x); x = dpp_add<0x4E>(x);
    x = dpp_add<0x141>(x); x = dpp_add<0x140>(x);
    return x;
}

// ---------------------------------------------------------------------------
// Forward (denominator) kernel. One wave owns 16 batch rows for the whole
// sequence. alpha is kept in WMMA D layout (two 16x16 f32 tiles: tags 0-15,
// 16-31 with pad 25..31 = NEGBIG). Per step:
//   m      = rowwise max(alpha)               (DPP butterfly)
//   p      = exp2((alpha-m)*log2e) as f16     -> LDS [row][tag]
//   A      = p reloaded in WMMA A layout      (2x ds_load_b128)
//   s      = A @ E via v_wmma_f32_16x16x32_f16 (E = exp(trans), const regs)
//   alpha' = alpha + f*(m + ln2*log2(s) + emit - alpha),  f = mask in {0,1}
// (branch-free masked update; pad tags handled by zero B columns + clamp).
// Emissions for step t+1 are prefetched during step t.
// ---------------------------------------------------------------------------
__global__ __launch_bounds__(32 * WAVES) void crf_forward_kernel(
    const float* __restrict__ emissions,
    const float* __restrict__ start_tr,
    const float* __restrict__ end_tr,
    const float* __restrict__ transitions,
    const unsigned char* __restrict__ mask,
    float* __restrict__ denom_out)
{
    __shared__ __align__(16) _Float16 lds[WAVES][16][32];

    const int tid   = threadIdx.x;
    const int wave  = tid >> 5;
    const int lane  = tid & 31;
    const int n     = lane & 15;       // tag column within a 16-wide tile
    const int hi    = lane >> 4;       // 0: lanes 0-15, 1: lanes 16-31
    const int row0  = (blockIdx.x * WAVES + wave) * 16;  // first batch row
    const int t1    = 16 + n;          // tag index in second tile
    const int t1c   = (t1 < NTAG) ? t1 : (NTAG - 1);     // clamped for loads

    // ---- B tiles: E = exp(transitions), padded to 32x32, WMMA B layout ----
    // lane n holds column n; lanes 0-15 carry K=0..15, lanes 16-31 K=16..31.
    v16h bm0, bm1;
    {
        const int kbase = hi ? 16 : 0;
#pragma unroll
        for (int h = 0; h < 16; ++h) {
            int k = kbase + h;
            float e0 = (k < NTAG) ? __expf(transitions[k * NTAG + n]) : 0.0f;
            float e1 = (k < NTAG && t1 < NTAG) ? __expf(transitions[k * NTAG + t1]) : 0.0f;
            bm0[h] = (_Float16)e0;
            bm1[h] = (_Float16)e1;
        }
    }

    // ---- per-row sequence lengths (mask is a prefix of ones) ----
    int lenMine = 0;
    {
        const unsigned* mrow = (const unsigned*)(mask + (size_t)(row0 + n) * SEQ);
#pragma unroll 4
        for (int i = 0; i < SEQ / 4; ++i) {
            unsigned w = mrow[i];
            lenMine += (int)((w * 0x01010101u) >> 24);   // sum of 4 bytes (0/1)
        }
    }
    int len_r[8];
#pragma unroll
    for (int r = 0; r < 8; ++r)
        len_r[r] = __shfl(lenMine, r + hi * 8, 32);

    const size_t rowStride = (size_t)SEQ * NTAG;

    // ---- alpha0 = start + emissions[:,0,:]  (D layout) ----
    float a0[8], a1[8];
#pragma unroll
    for (int r = 0; r < 8; ++r) {
        const float* em = emissions + (size_t)(row0 + r + hi * 8) * rowStride;
        a0[r] = start_tr[n] + em[n];
        a1[r] = (t1 < NTAG) ? (start_tr[t1] + em[t1]) : NEGBIG;
    }

    // Emission base pointers at t=1 (all 16 loads use 2 address regs +
    // immediate row-stride offsets).
    const float* p0base = emissions + (size_t)(row0 + hi * 8) * rowStride + NTAG + n;
    const float* p1base = emissions + (size_t)(row0 + hi * 8) * rowStride + NTAG + t1c;

    // Preload emissions for t = 1.
    float e0c[8], e1c[8];
#pragma unroll
    for (int r = 0; r < 8; ++r) {
        e0c[r] = p0base[r * rowStride];
        e1c[r] = p1base[r * rowStride];
    }

    const int rowM = n;   // A-matrix row owned by this lane
    const int sel  = hi;  // K-group select for A layout

    for (int t = 1; t < SEQ; ++t) {
        // ---- prefetch emissions for t+1 (clamped bump on last iter) ----
        int bump = (t < SEQ - 1) ? NTAG : 0;
        p0base += bump; p1base += bump;
        float e0n[8], e1n[8];
#pragma unroll
        for (int r = 0; r < 8; ++r) {
            e0n[r] = p0base[r * rowStride];
            e1n[r] = p1base[r * rowStride];
        }

        // ---- rowwise max over tags (DPP butterfly per 16-lane group) ----
        float m_r[8];
#pragma unroll
        for (int r = 0; r < 8; ++r)
            m_r[r] = red16_max(fmaxf(a0[r], a1[r]));

        // ---- p = exp2((alpha-m)*log2e) as f16 -> LDS [row][tag] ----
#pragma unroll
        for (int r = 0; r < 8; ++r) {
            int row = r + hi * 8;
            float ms = -m_r[r] * LOG2E;
            lds[wave][row][n]      = (_Float16)__builtin_amdgcn_exp2f(fmaf(a0[r], LOG2E, ms));
            lds[wave][row][16 + n] = (_Float16)__builtin_amdgcn_exp2f(fmaf(a1[r], LOG2E, ms));
        }

        // A layout: lane<16 -> K 0..15 of its row; lane>=16 -> K 16..31
        v8h lo = *(const v8h*)&lds[wave][rowM][sel * 16];
        v8h hv = *(const v8h*)&lds[wave][rowM][sel * 16 + 8];
        v16h amat = __builtin_shufflevector(lo, hv, 0, 1, 2, 3, 4, 5, 6, 7,
                                                    8, 9, 10, 11, 12, 13, 14, 15);
        v8f zero = {};
        v8f d0 = __builtin_amdgcn_wmma_f32_16x16x32_f16(
            false, amat, false, bm0, (short)0, zero, false, false);
        v8f d1 = __builtin_amdgcn_wmma_f32_16x16x32_f16(
            false, amat, false, bm1, (short)0, zero, false, false);

        // ---- alpha' = alpha + f*(m + ln2*log2(s) + emit - alpha) ----
        // f in {0,1}: branch-free masked update (fma is not if-convertible).
        // pad tags: d1 == 0 -> log2 = -inf -> clamped back to NEGBIG.
#pragma unroll
        for (int r = 0; r < 8; ++r) {
            float f   = (t < len_r[r]) ? 1.0f : 0.0f;
            float na0 = fmaf(__builtin_amdgcn_logf(d0[r]), LN2, m_r[r] + e0c[r]);
            float na1 = fmaf(__builtin_amdgcn_logf(d1[r]), LN2, m_r[r] + e1c[r]);
            na1 = fmaxf(na1, NEGBIG);
            a0[r] = fmaf(f, na0 - a0[r], a0[r]);
            a1[r] = fmaf(f, na1 - a1[r], a1[r]);
            e0c[r] = e0n[r];
            e1c[r] = e1n[r];
        }
    }

    // ---- denominator = logsumexp(alpha + end_transitions) per row ----
#pragma unroll
    for (int r = 0; r < 8; ++r) {
        float v0 = a0[r] + end_tr[n];
        float v1 = (t1 < NTAG) ? (a1[r] + end_tr[t1]) : NEGBIG;
        float mv = red16_max(fmaxf(v0, v1));
        float sv = __builtin_amdgcn_exp2f((v0 - mv) * LOG2E) +
                   __builtin_amdgcn_exp2f((v1 - mv) * LOG2E);
        sv = red16_add(sv);
        if (n == 0)
            denom_out[row0 + r + hi * 8] = fmaf(__builtin_amdgcn_logf(sv), LN2, mv);
    }
}

// ---------------------------------------------------------------------------
// Numerator (gold-path score) + mask count: one thread per batch row.
// ---------------------------------------------------------------------------
__global__ __launch_bounds__(256) void crf_numer_kernel(
    const float* __restrict__ emissions,
    const float* __restrict__ start_tr,
    const float* __restrict__ end_tr,
    const float* __restrict__ transitions,
    const int* __restrict__ labels,
    const unsigned char* __restrict__ mask,
    float* __restrict__ num_out,
    float* __restrict__ cnt_out)
{
    int b = blockIdx.x * blockDim.x + threadIdx.x;
    if (b >= BATCH) return;
    const int*           lb = labels + (size_t)b * SEQ;
    const unsigned char* mk = mask   + (size_t)b * SEQ;
    const float*         em = emissions + (size_t)b * SEQ * NTAG;

    int   prev  = lb[0];
    int   last  = prev;
    int   len   = 1;
    float score = start_tr[prev] + em[prev];
    for (int t = 1; t < SEQ; ++t) {
        int cur = lb[t];
        if (mk[t]) {  // mask is a prefix of ones
            score += transitions[prev * NTAG + cur] + em[t * NTAG + cur];
            last = cur;
            ++len;
        }
        prev = cur;
    }
    num_out[b] = score + end_tr[last];
    cnt_out[b] = (float)len;
}

// ---------------------------------------------------------------------------
// Deterministic final reduction: sum(num - den) / sum(cnt)
// ---------------------------------------------------------------------------
__global__ __launch_bounds__(1024) void crf_reduce_kernel(
    const float* __restrict__ num,
    const float* __restrict__ den,
    const float* __restrict__ cnt,
    float* __restrict__ out)
{
    __shared__ float sl[1024];
    __shared__ float sc[1024];
    int tid = threadIdx.x;
    float s = 0.0f, c = 0.0f;
    for (int b = tid; b < BATCH; b += 1024) {
        s += num[b] - den[b];
        c += cnt[b];
    }
    sl[tid] = s; sc[tid] = c;
    __syncthreads();
    for (int off = 512; off > 0; off >>= 1) {
        if (tid < off) { sl[tid] += sl[tid + off]; sc[tid] += sc[tid + off]; }
        __syncthreads();
    }
    if (tid == 0) out[0] = sl[0] / sc[0];
}

extern "C" void kernel_launch(void* const* d_in, const int* in_sizes, int n_in,
                              void* d_out, int out_size, void* d_ws, size_t ws_size,
                              hipStream_t stream)
{
    const float*         emissions = (const float*)d_in[0];
    const float*         start_tr  = (const float*)d_in[1];
    const float*         end_tr    = (const float*)d_in[2];
    const float*         trans     = (const float*)d_in[3];
    const int*           labels    = (const int*)d_in[4];
    const unsigned char* mask      = (const unsigned char*)d_in[5];

    float* ws  = (float*)d_ws;
    float* den = ws;              // BATCH floats
    float* num = ws + BATCH;      // BATCH floats
    float* cnt = ws + 2 * BATCH;  // BATCH floats

    // 256 waves total (16 batch rows per wave), WAVES waves per block
    crf_forward_kernel<<<BATCH / (16 * WAVES), 32 * WAVES, 0, stream>>>(
        emissions, start_tr, end_tr, trans, mask, den);
    crf_numer_kernel<<<BATCH / 256, 256, 0, stream>>>(
        emissions, start_tr, end_tr, trans, labels, mask, num, cnt);
    crf_reduce_kernel<<<1, 1024, 0, stream>>>(num, den, cnt, (float*)d_out);
}
